// MultilaterationHead_66168266162350
// MI455X (gfx1250) — matrix operand order repaired and checked by reading, not compiled, not run on hardware
//
#include <hip/hip_runtime.h>
#include <hip/hip_bf16.h>

typedef __attribute__((ext_vector_type(16))) _Float16 v16h;
typedef __attribute__((ext_vector_type(8)))  _Float16 v8h;
typedef __attribute__((ext_vector_type(8)))  float    v8f;

#define EMB   512
#define H1    300
#define H1P   320   // padded K (10 chunks of 32)
#define H2    100
#define H2P   112   // padded cols (7 tiles of 16)
#define NPTS  256
#define BATCH 2
#define NKC   10    // H1P/32
#define NT    7     // H2P/16

// ---------------------------------------------------------------------------
// init: zero accumulators, build B-fragment-swizzled f16 W2, padded b2/W3.
// W2frag layout: [(kc*7 + t)*32 + lane]*16 + j  (contiguous 16 halves/lane)
//   lane<16 : col = t*16+lane,    K = kc*32 + (j<8 ? j : 16+j-8)
//   lane>=16: col = t*16+lane-16, K = kc*32 + 8 + (j<8 ? j : 16+j-8)
// ---------------------------------------------------------------------------
__global__ void init_ws_kernel(const float* __restrict__ W2,
                               const float* __restrict__ b2,
                               const float* __restrict__ W3,
                               _Float16* __restrict__ W2frag,
                               float* __restrict__ b2p,
                               float* __restrict__ W3p,
                               float* __restrict__ a_acc,
                               float* __restrict__ mwd_acc) {
  int idx = blockIdx.x * blockDim.x + threadIdx.x;
  const int nFrag = NKC * NT * 32 * 16;            // 35840
  if (idx < nFrag) {
    const int j    = idx & 15;
    const int lane = (idx >> 4) & 31;
    const int t    = (idx >> 9) % NT;
    const int kc   = idx / (16 * 32 * NT);
    const int c    = t * 16 + (lane & 15);
    const int k    = kc * 32 + (lane >= 16 ? 8 : 0) + (j & 7) + ((j & 8) ? 16 : 0);
    float v = (k < H1 && c < H2) ? W2[k * H2 + c] : 0.0f;
    W2frag[idx] = (_Float16)v;
    return;
  }
  idx -= nFrag;
  if (idx < H2P) { b2p[idx] = (idx < H2) ? b2[idx] : 0.0f; return; }
  idx -= H2P;
  if (idx < H2P) { W3p[idx] = (idx < H2) ? W3[idx] : 0.0f; return; }  // W3 is (100,1)
  idx -= H2P;
  if (idx < BATCH * NPTS * 3) { a_acc[idx] = 0.0f; return; }
  idx -= BATCH * NPTS * 3;
  if (idx < BATCH * NPTS) { mwd_acc[idx] = 0.0f; return; }
}

// ---------------------------------------------------------------------------
// stage A: layer-1 GEMMs in fp32, stored as f16 with K padded to 320.
//   hAa[b][n][k] = sum_f embA[b][f][n]*W1[f][k]       + b1[k]   (b1 folded)
//   hAb[b][n][k] = sum_f embA[b][f][n]*W1[512+f][k]
//   hBa[b][m][k] = sum_f embB[b][f][m]*W1[f][k]       + b1[k]   (b1 folded)
//   hBb[b][m][k] = sum_f embB[b][f][m]*W1[512+f][k]
// ---------------------------------------------------------------------------
__global__ void stage_a_kernel(const float* __restrict__ embA,
                               const float* __restrict__ embB,
                               const float* __restrict__ W1,
                               const float* __restrict__ b1,
                               _Float16* __restrict__ hAa,
                               _Float16* __restrict__ hAb,
                               _Float16* __restrict__ hBa,
                               _Float16* __restrict__ hBb) {
  int idx = blockIdx.x * blockDim.x + threadIdx.x;
  const int per = BATCH * NPTS * H1P;
  if (idx >= 4 * per) return;
  int mat = idx / per;
  int rem = idx % per;
  int b   = rem / (NPTS * H1P);
  int r2  = rem % (NPTS * H1P);
  int row = r2 / H1P;
  int k   = r2 % H1P;

  _Float16* outp = hAa;
  const float* E = embA;
  int wofs = 0;
  bool addb1 = false;
  if (mat == 0)      { outp = hAa; E = embA; wofs = 0;        addb1 = true;  }
  else if (mat == 1) { outp = hAb; E = embA; wofs = EMB * H1; addb1 = false; }
  else if (mat == 2) { outp = hBa; E = embB; wofs = 0;        addb1 = true;  }
  else               { outp = hBb; E = embB; wofs = EMB * H1; addb1 = false; }

  float v = 0.0f;
  if (k < H1) {
    float acc = 0.0f;
    const float* e = E + (size_t)b * EMB * NPTS + row;
    const float* w = W1 + wofs + k;
#pragma unroll 4
    for (int f = 0; f < EMB; ++f)
      acc += e[(size_t)f * NPTS] * w[(size_t)f * H1];
    if (addb1) acc += b1[k];
    v = acc;
  }
  outp[(size_t)b * NPTS * H1P + (size_t)row * H1P + k] = (_Float16)v;
}

// ---------------------------------------------------------------------------
// stage B: pairwise MLP via WMMA.  Block = one (16n x 16m) tile of one batch.
// Rows 0..255 = v1 pairs, rows 256..511 = v2 pairs of the same GEMM
//   X(row,k) = relu(h0[k] + h1[k]) ,  Y = X * W2 (320x112, f16, f32 accum)
// Then +b2, relu, dot W3, softplus, d = |P_m|^2 - R^2, reduce over m.
// ---------------------------------------------------------------------------
__launch_bounds__(256)
__global__ void stage_b_kernel(const _Float16* __restrict__ hAa,
                               const _Float16* __restrict__ hAb,
                               const _Float16* __restrict__ hBa,
                               const _Float16* __restrict__ hBb,
                               const _Float16* __restrict__ W2frag,
                               const float* __restrict__ b2p,
                               const float* __restrict__ W3p,
                               const float* __restrict__ b3,
                               const float* __restrict__ anchor_points,
                               float* __restrict__ a_acc,
                               float* __restrict__ mwd_acc) {
  __shared__ _Float16 tAa[16 * H1P];
  __shared__ _Float16 tAb[16 * H1P];
  __shared__ _Float16 tBa[16 * H1P];
  __shared__ _Float16 tBb[16 * H1P];
  __shared__ float outAcc[512];
  __shared__ float aPart[16 * 3];
  __shared__ float mPart[16];

  const int nT  = blockIdx.x;
  const int mT  = blockIdx.y;
  const int b   = blockIdx.z;
  const int tid = threadIdx.x;

  // stage h tiles to LDS (128-bit vector copies: 8 halves per op)
  {
    const v8h* gAa = (const v8h*)(hAa + (size_t)b * NPTS * H1P + (size_t)nT * 16 * H1P);
    const v8h* gAb = (const v8h*)(hAb + (size_t)b * NPTS * H1P + (size_t)nT * 16 * H1P);
    const v8h* gBa = (const v8h*)(hBa + (size_t)b * NPTS * H1P + (size_t)mT * 16 * H1P);
    const v8h* gBb = (const v8h*)(hBb + (size_t)b * NPTS * H1P + (size_t)mT * 16 * H1P);
    v8h* sAa = (v8h*)tAa; v8h* sAb = (v8h*)tAb;
    v8h* sBa = (v8h*)tBa; v8h* sBb = (v8h*)tBb;
    const int nv = 16 * H1P / 8;                  // 640 vectors
    for (int i = tid; i < nv; i += 256) {
      sAa[i] = gAa[i];
      sAb[i] = gAb[i];
      sBa[i] = gBa[i];
      sBb[i] = gBb[i];
    }
  }
  for (int i = tid; i < 512; i += 256) outAcc[i] = 0.0f;
  if (tid < 16) {
    mPart[tid] = 0.0f;
    aPart[tid * 3 + 0] = 0.0f;
    aPart[tid * 3 + 1] = 0.0f;
    aPart[tid * 3 + 2] = 0.0f;
  }
  __syncthreads();

  const int  lane   = tid & 31;
  const int  wave   = tid >> 5;        // 8 waves per block
  const int  rowf   = lane & 15;       // A-matrix row for this lane
  const bool hiHalf = lane >= 16;
  const int  klo    = hiHalf ? 8 : 0;  // A layout: K-half select
  const int  col    = lane & 15;       // B / D column for this lane

  // lane-constant epilogue constants (hoisted)
  float b2r[NT], w3r[NT];
#pragma unroll
  for (int t = 0; t < NT; ++t) {
    b2r[t] = b2p[t * 16 + col];
    w3r[t] = W3p[t * 16 + col];
  }

  for (int fi = wave; fi < 32; fi += 8) {   // 32 row-fragments of 16 pairs
    const int p = fi * 16 + rowf;           // global row id (0..511)
    const _Float16* x0;
    const _Float16* x1;
    if (p < 256) {                          // v1: relu(hAa[n] + hBb[m])
      x0 = &tAa[(p >> 4) * H1P];
      x1 = &tBb[(p & 15) * H1P];
    } else {                                // v2: relu(hBa[m] + hAb[n])
      const int q = p - 256;
      x0 = &tBa[(q & 15) * H1P];
      x1 = &tAb[(q >> 4) * H1P];
    }

    v8f acc[NT];
#pragma unroll
    for (int t = 0; t < NT; ++t) acc[t] = (v8f){0, 0, 0, 0, 0, 0, 0, 0};

    for (int kc = 0; kc < NKC; ++kc) {
      const int k0 = kc * 32;
      // A fragment: two contiguous 8-half runs, packed f16 add + relu
      const v8h a0 = *(const v8h*)(x0 + k0 + klo);
      const v8h b0 = *(const v8h*)(x1 + k0 + klo);
      const v8h a1 = *(const v8h*)(x0 + k0 + 16 + klo);
      const v8h b1v = *(const v8h*)(x1 + k0 + 16 + klo);
      v8h slo = a0 + b0;
      v8h shi = a1 + b1v;
      const v8h zero = (v8h)(_Float16)0;
      slo = __builtin_elementwise_max(slo, zero);
      shi = __builtin_elementwise_max(shi, zero);
      const v16h afrag = __builtin_shufflevector(
          slo, shi, 0, 1, 2, 3, 4, 5, 6, 7, 8, 9, 10, 11, 12, 13, 14, 15);

#pragma unroll
      for (int t = 0; t < NT; ++t) {
        // pre-swizzled B fragment: one aligned 32-byte load per lane
        const v16h bfrag = *(const v16h*)(
            W2frag + (((size_t)(kc * NT + t)) * 32 + lane) * 16);
        acc[t] = __builtin_amdgcn_wmma_f32_16x16x32_f16(
            false, afrag, false, bfrag, (short)0, acc[t], false, false);
      }
    }

    // epilogue: +b2, relu, dot with W3 -> per-pair scalar in LDS
#pragma unroll
    for (int t = 0; t < NT; ++t) {
#pragma unroll
      for (int r = 0; r < 8; ++r) {
        const int prow = fi * 16 + r + (hiHalf ? 8 : 0);
        float g = acc[t][r] + b2r[t];
        g = g > 0.0f ? g : 0.0f;
        atomicAdd(&outAcc[prow], g * w3r[t]);
      }
    }
  }
  __syncthreads();

  // per-pair finalize: softplus, d-value, tile reductions over m
  {
    const int p  = tid;                     // 0..255, pair (nl, ml)
    const int nl = p >> 4;
    const int ml = p & 15;
    const float bias3 = b3[0];
    const float v1 = outAcc[p] + bias3;
    const float v2 = outAcc[256 + p] + bias3;
    const float s  = 0.5f * (v1 + v2);
    const float R  = s > 0.0f ? s + log1pf(expf(-s)) : log1pf(expf(s));
    const int m = mT * 16 + ml;
    const float* ap = anchor_points + (size_t)b * 3 * NPTS;
    const float px = ap[0 * NPTS + m];
    const float py = ap[1 * NPTS + m];
    const float pz = ap[2 * NPTS + m];
    const float norm2 = px * px + py * py + pz * pz;
    const float dval = norm2 - R * R;
    atomicAdd(&mPart[nl], dval);
    atomicAdd(&aPart[nl * 3 + 0], dval * px);
    atomicAdd(&aPart[nl * 3 + 1], dval * py);
    atomicAdd(&aPart[nl * 3 + 2], dval * pz);
  }
  __syncthreads();

  if (tid < 64) {
    const int nl = tid >> 2, comp = tid & 3;
    const int n = nT * 16 + nl;
    if (comp == 3)
      atomicAdd(&mwd_acc[b * NPTS + n], mPart[nl]);
    else
      atomicAdd(&a_acc[(b * NPTS + n) * 3 + comp], aPart[nl * 3 + comp]);
  }
}

// ---------------------------------------------------------------------------
// stage D: multilateration closure per batch + flow output
// ---------------------------------------------------------------------------
__global__ void stage_d_kernel(const float* __restrict__ a_acc,
                               const float* __restrict__ mwd_acc,
                               const float* __restrict__ anchor_points,
                               const float* __restrict__ action_points,
                               float* __restrict__ out) {
  const int b = blockIdx.x;
  __shared__ float cS[3];
  __shared__ float HinvS[9];
  __shared__ float PmS[9];

  if (threadIdx.x == 0) {
    const float* P = anchor_points + (size_t)b * 3 * NPTS;
    float c0 = 0, c1 = 0, c2 = 0;
    float pp0 = 0, pp1 = 0, pp2 = 0, pp4 = 0, pp5 = 0, pp8 = 0;
    for (int m = 0; m < NPTS; ++m) {
      const float x = P[m], y = P[NPTS + m], z = P[2 * NPTS + m];
      c0 += x; c1 += y; c2 += z;
      pp0 += x * x; pp1 += x * y; pp2 += x * z;
      pp4 += y * y; pp5 += y * z; pp8 += z * z;
    }
    const float inv_m = 1.0f / (float)NPTS;
    c0 *= inv_m; c1 *= inv_m; c2 *= inv_m;
    float Pm[9];
    Pm[0] = pp0 * inv_m; Pm[1] = pp1 * inv_m; Pm[2] = pp2 * inv_m;
    Pm[3] = Pm[1];       Pm[4] = pp4 * inv_m; Pm[5] = pp5 * inv_m;
    Pm[6] = Pm[2];       Pm[7] = Pm[5];       Pm[8] = pp8 * inv_m;
    const float cv[3] = {c0, c1, c2};
    float H[9];
    for (int i = 0; i < 3; ++i)
      for (int j = 0; j < 3; ++j)
        H[i * 3 + j] = -2.0f * Pm[i * 3 + j] + 2.0f * cv[i] * cv[j];
    const float det = H[0] * (H[4] * H[8] - H[5] * H[7])
                    - H[1] * (H[3] * H[8] - H[5] * H[6])
                    + H[2] * (H[3] * H[7] - H[4] * H[6]);
    const float id = 1.0f / det;
    HinvS[0] = (H[4] * H[8] - H[5] * H[7]) * id;
    HinvS[1] = (H[2] * H[7] - H[1] * H[8]) * id;
    HinvS[2] = (H[1] * H[5] - H[2] * H[4]) * id;
    HinvS[3] = (H[5] * H[6] - H[3] * H[8]) * id;
    HinvS[4] = (H[0] * H[8] - H[2] * H[6]) * id;
    HinvS[5] = (H[2] * H[3] - H[0] * H[5]) * id;
    HinvS[6] = (H[3] * H[7] - H[4] * H[6]) * id;
    HinvS[7] = (H[1] * H[6] - H[0] * H[7]) * id;
    HinvS[8] = (H[0] * H[4] - H[1] * H[3]) * id;
    cS[0] = c0; cS[1] = c1; cS[2] = c2;
    for (int i = 0; i < 9; ++i) PmS[i] = Pm[i];
  }
  __syncthreads();

  const int n = threadIdx.x;  // 0..255
  const float c0 = cS[0], c1 = cS[1], c2 = cS[2];
  const float cc = c0 * c0 + c1 * c1 + c2 * c2;
  const float inv_m = 1.0f / (float)NPTS;
  const float a0 = a_acc[(b * NPTS + n) * 3 + 0] * inv_m;
  const float a1 = a_acc[(b * NPTS + n) * 3 + 1] * inv_m;
  const float a2 = a_acc[(b * NPTS + n) * 3 + 2] * inv_m;
  const float mwd = mwd_acc[b * NPTS + n] * inv_m;
  // WPPt == PPt_mean (W = ones)
  const float Pc0 = PmS[0] * c0 + PmS[1] * c1 + PmS[2] * c2;
  const float Pc1 = PmS[3] * c0 + PmS[4] * c1 + PmS[5] * c2;
  const float Pc2 = PmS[6] * c0 + PmS[7] * c1 + PmS[8] * c2;
  const float f0 = a0 - 2.0f * Pc0 - mwd * c0 + 2.0f * cc * c0;
  const float f1 = a1 - 2.0f * Pc1 - mwd * c1 + 2.0f * cc * c1;
  const float f2 = a2 - 2.0f * Pc2 - mwd * c2 + 2.0f * cc * c2;
  const float q0 = -(HinvS[0] * f0 + HinvS[1] * f1 + HinvS[2] * f2);
  const float q1 = -(HinvS[3] * f0 + HinvS[4] * f1 + HinvS[5] * f2);
  const float q2 = -(HinvS[6] * f0 + HinvS[7] * f1 + HinvS[8] * f2);
  const float* PA = action_points + (size_t)b * 3 * NPTS;
  out[(size_t)b * 3 * NPTS + 0 * NPTS + n] = (q0 + c0) - PA[0 * NPTS + n];
  out[(size_t)b * 3 * NPTS + 1 * NPTS + n] = (q1 + c1) - PA[1 * NPTS + n];
  out[(size_t)b * 3 * NPTS + 2 * NPTS + n] = (q2 + c2) - PA[2 * NPTS + n];
}

// ---------------------------------------------------------------------------
extern "C" void kernel_launch(void* const* d_in, const int* in_sizes, int n_in,
                              void* d_out, int out_size, void* d_ws, size_t ws_size,
                              hipStream_t stream) {
  const float* embA = (const float*)d_in[0];  // (B,512,256)
  const float* embB = (const float*)d_in[1];  // (B,512,256)
  const float* actP = (const float*)d_in[2];  // (B,3,256)
  const float* ancP = (const float*)d_in[3];  // (B,3,256)
  const float* W1   = (const float*)d_in[4];  // (1024,300)
  const float* b1   = (const float*)d_in[5];  // (300,)
  const float* W2   = (const float*)d_in[6];  // (300,100)
  const float* b2   = (const float*)d_in[7];  // (100,)
  const float* W3   = (const float*)d_in[8];  // (100,1)
  const float* b3   = (const float*)d_in[9];  // (1,)
  float* out = (float*)d_out;

  char* ws = (char*)d_ws;
  size_t off = 0;
  auto alloc = [&](size_t bytes) -> void* {
    void* p = ws + off;
    off += (bytes + 255) & ~(size_t)255;
    return p;
  };
  const size_t hBytes = (size_t)BATCH * NPTS * H1P * sizeof(_Float16);
  _Float16* hAa    = (_Float16*)alloc(hBytes);
  _Float16* hAb    = (_Float16*)alloc(hBytes);
  _Float16* hBa    = (_Float16*)alloc(hBytes);
  _Float16* hBb    = (_Float16*)alloc(hBytes);
  _Float16* W2frag = (_Float16*)alloc((size_t)NKC * NT * 32 * 16 * sizeof(_Float16));
  float* b2p    = (float*)alloc(H2P * sizeof(float));
  float* W3p    = (float*)alloc(H2P * sizeof(float));
  float* a_acc  = (float*)alloc((size_t)BATCH * NPTS * 3 * sizeof(float));
  float* mwd    = (float*)alloc((size_t)BATCH * NPTS * sizeof(float));

  {
    const int total = NKC * NT * 32 * 16 + H2P + H2P + BATCH * NPTS * 3 + BATCH * NPTS;
    init_ws_kernel<<<(total + 255) / 256, 256, 0, stream>>>(
        W2, b2, W3, W2frag, b2p, W3p, a_acc, mwd);
  }
  {
    const int total = 4 * BATCH * NPTS * H1P;
    stage_a_kernel<<<(total + 255) / 256, 256, 0, stream>>>(
        embA, embB, W1, b1, hAa, hAb, hBa, hBb);
  }
  {
    dim3 grid(NPTS / 16, NPTS / 16, BATCH);   // (16,16,2)
    stage_b_kernel<<<grid, 256, 0, stream>>>(
        hAa, hAb, hBa, hBb, W2frag, b2p, W3p, b3, ancP, a_acc, mwd);
  }
  stage_d_kernel<<<BATCH, NPTS, 0, stream>>>(a_acc, mwd, ancP, actP, out);
}
